// SCMMContextTelMESC_7318624272749
// MI455X (gfx1250) — matrix operand order repaired and checked by reading, not compiled, not run on hardware
//
#include <hip/hip_runtime.h>
#include <hip/hip_bf16.h>
#include <stdint.h>

// ---------------------------------------------------------------------------
// Sizes (fixed by the reference)
// ---------------------------------------------------------------------------
constexpr int kB   = 8192;
constexpr int kH   = 768;
constexpr int kS   = 12;
constexpr int kNH  = 8;
constexpr int kHD  = 96;
constexpr int kFF  = 2048;
constexpr int kNROW = kB * kS;         // 98304 rows for the big GEMMs
constexpr int kQKVN = 3 * kH;          // 2304

typedef _Float16 half_t;
typedef __attribute__((ext_vector_type(16))) _Float16 v16h;
typedef __attribute__((ext_vector_type(8)))  _Float16 v8h;
typedef __attribute__((ext_vector_type(8)))  float    v8f;
typedef __attribute__((ext_vector_type(4)))  unsigned int u32x4;
typedef __attribute__((ext_vector_type(4)))  int      i32x4;
typedef __attribute__((ext_vector_type(8)))  int      i32x8;
// GCC-style vector for async-to-LDS builtin pointer params (must match exactly)
typedef int gccv4i __attribute__((vector_size(16)));

#define HAS_TDM        __has_builtin(__builtin_amdgcn_tensor_load_to_lds)
#define HAS_ASYNC_LDS  __has_builtin(__builtin_amdgcn_global_load_async_to_lds_b128)

__device__ __forceinline__ float wave_red_sum(float v) {
  #pragma unroll
  for (int m = 16; m > 0; m >>= 1) v += __shfl_xor(v, m, 32);
  return v;
}

// ---------------------------------------------------------------------------
// CDNA5 data-movement helpers
// ---------------------------------------------------------------------------
#if HAS_TDM
// One TDM descriptor: 2D tile, 128 rows x 32 f16 (64B), global row stride = K
// elements, LDS padded to 40-half (80B) rows via pad_interval=16DW/pad_amount=4DW.
__device__ __forceinline__ void tdm_load_tile_128x32(const half_t* gsrc,
                                                     half_t* ldst, int K) {
  const uint64_t ga = (uint64_t)(uintptr_t)gsrc;
  const uint32_t la = (uint32_t)(uintptr_t)ldst;   // flat low 32 bits == LDS offset
  u32x4 g0;
  g0[0] = 1u;                                       // count=1 (valid), no gather
  g0[1] = la;                                       // lds_addr
  g0[2] = (uint32_t)ga;                             // global_addr[31:0]
  g0[3] = (uint32_t)((ga >> 32) & 0x01FFFFFFu) | (2u << 30);  // addr[56:32], type=2
  i32x8 g1;
  g1[0] = (1 << 16) | (1 << 20) | (3 << 22) | (3 << 25); // data_size=2B, pad_en,
                                                         // interval=16DW, amount=4DW
  g1[1] = (int)(((unsigned)K & 0xFFFFu) << 16);     // tensor_dim0 lo16
  g1[2] = (int)((((unsigned)K >> 16) & 0xFFFFu) | (128u << 16)); // dim0 hi16 | dim1 lo16
  g1[3] = (int)(32u << 16);                         // tile_dim0 = 32 elements
  g1[4] = 128;                                      // tile_dim1 = 128 rows
  g1[5] = K;                                        // tensor_dim0_stride (lo32)
  g1[6] = 0;
  g1[7] = 0;
  const i32x4 z4 = {0, 0, 0, 0};
#if defined(__clang_major__) && (__clang_major__ >= 23)
  const i32x8 z8 = {0, 0, 0, 0, 0, 0, 0, 0};
  __builtin_amdgcn_tensor_load_to_lds(g0, g1, z4, z4, z8, 0);
#else
  __builtin_amdgcn_tensor_load_to_lds(g0, g1, z4, z4, 0);
#endif
}
#endif

__device__ __forceinline__ void wait_tensorcnt0() {
#if __has_builtin(__builtin_amdgcn_s_wait_tensorcnt)
  __builtin_amdgcn_s_wait_tensorcnt(0);
#else
  asm volatile("s_wait_tensorcnt 0x0" ::: "memory");
#endif
}

__device__ __forceinline__ void wait_asynccnt0() {
#if __has_builtin(__builtin_amdgcn_s_wait_asynccnt)
  __builtin_amdgcn_s_wait_asynccnt(0);
#else
  asm volatile("s_wait_asynccnt 0x0" ::: "memory");
#endif
}

__device__ __forceinline__ void copy16_to_lds(const half_t* g, half_t* l) {
#if HAS_ASYNC_LDS
  __builtin_amdgcn_global_load_async_to_lds_b128(
      (__attribute__((address_space(1))) gccv4i*)(uintptr_t)g,
      (__attribute__((address_space(3))) gccv4i*)(uintptr_t)l, 0, 0);
#else
  *reinterpret_cast<v8h*>(l) = *reinterpret_cast<const v8h*>(g);
#endif
}

// ---------------------------------------------------------------------------
// Weight conversion: f32 [K][N] -> f16 transposed [N][K]
// ---------------------------------------------------------------------------
__global__ void cvt_transpose_kernel(const float* __restrict__ s, half_t* __restrict__ d,
                                     int K, int N) {
  int i = blockIdx.x * 256 + threadIdx.x;          // over N*K, d is [N][K]
  if (i >= K * N) return;
  int n = i / K, k = i % K;
  d[i] = (half_t)s[(size_t)k * N + n];
}

__global__ void pack_qkv_t_kernel(const float* __restrict__ Wq, const float* __restrict__ Wk,
                                  const float* __restrict__ Wv, half_t* __restrict__ out) {
  int i = blockIdx.x * 256 + threadIdx.x;          // out is [2304][768]
  if (i >= kQKVN * kH) return;
  int n = i / kH, k = i % kH;
  const float* src = (n < kH) ? Wq : (n < 2 * kH) ? Wk : Wv;
  int nn = (n < kH) ? n : (n < 2 * kH) ? n - kH : n - 2 * kH;
  out[i] = (half_t)src[(size_t)k * kH + nn];
}

__global__ void pack_bias3_kernel(const float* __restrict__ bq, const float* __restrict__ bk,
                                  const float* __restrict__ bv, float* __restrict__ out) {
  int i = blockIdx.x * 256 + threadIdx.x;
  if (i >= kQKVN) return;
  out[i] = (i < kH) ? bq[i] : (i < 2 * kH) ? bk[i - kH] : bv[i - 2 * kH];
}

// ---------------------------------------------------------------------------
// Sequence builder: age-sorted gather + current token, local pool, lengths.
// ---------------------------------------------------------------------------
__global__ __launch_bounds__(256) void build_seq_kernel(
    const float* __restrict__ z, const float* __restrict__ buf,
    const int* __restrict__ bmask, const int* __restrict__ bidx,
    half_t* __restrict__ gseq, half_t* __restrict__ z16,
    half_t* __restrict__ lpool, int* __restrict__ glen) {
  const int b = blockIdx.x * 8 + (threadIdx.x >> 5);
  const int lane = threadIdx.x & 31;
  if (b >= kB) return;

  int key[12];
  const int idx = bidx[b];
  int nv = 0;
  #pragma unroll
  for (int j = 0; j < 12; ++j) {
    int m = bmask[b * 12 + j];
    int age = (idx - j + 24) % 12;
    key[j] = m ? (-age) : 100;          // oldest (largest age) first
    nv += m ? 1 : 0;
  }
  int ord[11];
  #pragma unroll
  for (int rank = 0; rank < 11; ++rank) {
    int bk = 1000, bj = 0;
    #pragma unroll
    for (int j = 0; j < 12; ++j)
      if (key[j] < bk) { bk = key[j]; bj = j; }
    ord[rank] = bj;
    key[bj] = 500 + rank;
  }
  const int hlen = nv < 11 ? nv : 11;
  const int hl4  = nv < 4 ? nv : 4;
  if (lane == 0) glen[b] = hlen + 1;    // reference quirk: row `hlen` counts as valid

  const float* zb = z + (size_t)b * kH;
  const float* bb = buf + (size_t)b * 12 * kH;
  half_t* gs = gseq + (size_t)b * kS * kH;
  #pragma unroll 4
  for (int t = 0; t < kH / 32; ++t) {
    const int c = lane + 32 * t;
    const float zc = zb[c];
    for (int s = 0; s < kS; ++s) {
      float v;
      if (s < hlen)      v = bb[(size_t)ord[s] * kH + c];
      else if (s == 11)  v = zc;
      else               v = 0.f;
      gs[(size_t)s * kH + c] = (half_t)v;
    }
    float lsum = 0.f;
    for (int s = 0; s < hl4; ++s) lsum += bb[(size_t)ord[s] * kH + c];
    if (hl4 == 4) lsum += zc;
    lpool[(size_t)b * kH + c] = (half_t)(lsum / (float)(hl4 + 1));
    z16[(size_t)b * kH + c] = (half_t)zc;
  }
}

// ---------------------------------------------------------------------------
// Tiled WMMA GEMM:  C[M,N] = act( A[M,K] @ Wt[N,K]^T + bias [+ resid] )
// f16 in, f32 accumulate, f16 out. Block tile 128x128, 8 waves of 32(M)x64(N),
// K-step 32, double-buffered LDS staged by TDM / async-to-LDS, 80B LDS rows.
// Compile-time: HR = add residual, ACT: 0=none 1=relu 2=tanh.
// ---------------------------------------------------------------------------
template <bool HR, int ACT>
__global__ __launch_bounds__(256) void gemm_f16_kernel(
    const half_t* __restrict__ A, const half_t* __restrict__ Wt,
    const float* __restrict__ bias, const half_t* __restrict__ resid,
    half_t* __restrict__ C, int M, int N, int K) {
  __shared__ half_t sA[2][128 * 40];
  __shared__ half_t sB[2][128 * 40];

  const int tid  = threadIdx.x;
  const int lane = tid & 31;
  const int wid  = tid >> 5;
  const int wm   = wid & 3;            // wave M group: 0..3  (x32 rows)
  const int wn   = wid >> 2;           // wave N group: 0..1  (x64 cols)
  const int m0   = blockIdx.y * 128;
  const int n0   = blockIdx.x * 128;
  const int nk   = K >> 5;

  v8f acc[2][4];
  #pragma unroll
  for (int mi = 0; mi < 2; ++mi)
    #pragma unroll
    for (int ni = 0; ni < 4; ++ni)
      #pragma unroll
      for (int r = 0; r < 8; ++r) acc[mi][ni][r] = 0.f;

  auto load_stage = [&](int st, int kt) {
#if HAS_TDM
    if (wid == 0) {                         // one TDM op per tile, wave-level
      tdm_load_tile_128x32(A  + (size_t)m0 * K + kt * 32, &sA[st][0], K);
      tdm_load_tile_128x32(Wt + (size_t)n0 * K + kt * 32, &sB[st][0], K);
    }
#else
    #pragma unroll
    for (int i = 0; i < 2; ++i) {           // A tile: 128 rows x 32 halves
      int idx = tid + i * 256;
      int row = idx >> 2, ch = idx & 3;
      copy16_to_lds(A + (size_t)(m0 + row) * K + kt * 32 + ch * 8,
                    &sA[st][row * 40 + ch * 8]);
    }
    #pragma unroll
    for (int i = 0; i < 2; ++i) {           // B tile: 128 out-chans x 32 halves
      int idx = tid + i * 256;
      int row = idx >> 2, ch = idx & 3;
      copy16_to_lds(Wt + (size_t)(n0 + row) * K + kt * 32 + ch * 8,
                    &sB[st][row * 40 + ch * 8]);
    }
#endif
    if (kt + 2 < nk)                        // global_prefetch_b8 on a future W tile
      __builtin_prefetch(Wt + (size_t)(n0 + (tid >> 1)) * K + (kt + 2) * 32 + (tid & 1) * 16, 0, 1);
  };

  auto sync_stage = [&]() {
#if HAS_TDM
    if (wid == 0) wait_tensorcnt0();
#elif HAS_ASYNC_LDS
    wait_asynccnt0();
#endif
    __syncthreads();
  };

  load_stage(0, 0);
  sync_stage();

  for (int kt = 0; kt < nk; ++kt) {
    const int cur = kt & 1;
    if (kt + 1 < nk) load_stage(cur ^ 1, kt + 1);

    // A fragments: lane<16 -> K {0..7,16..23}, lane>=16 -> K {8..15,24..31}
    v16h af[2];
    #pragma unroll
    for (int mi = 0; mi < 2; ++mi) {
      const int m  = wm * 32 + mi * 16 + (lane & 15);
      const int kh = (lane >> 4) * 8;
      const v8h lo = *reinterpret_cast<const v8h*>(&sA[cur][m * 40 + kh]);
      const v8h hi = *reinterpret_cast<const v8h*>(&sA[cur][m * 40 + 16 + kh]);
      v16h a;
      #pragma unroll
      for (int e = 0; e < 8; ++e) { a[e] = lo[e]; a[e + 8] = hi[e]; }
      af[mi] = a;
    }
    // B fragments: lane<16 -> K {0..7,16..23}, lane>=16 -> K {8..15,24..31}
    v16h bf[4];
    #pragma unroll
    for (int ni = 0; ni < 4; ++ni) {
      const int n  = wn * 64 + ni * 16 + (lane & 15);
      const int kb = (lane >> 4) * 8;
      const v8h lo = *reinterpret_cast<const v8h*>(&sB[cur][n * 40 + kb]);
      const v8h hi = *reinterpret_cast<const v8h*>(&sB[cur][n * 40 + 16 + kb]);
      v16h bvv;
      #pragma unroll
      for (int e = 0; e < 8; ++e) { bvv[e] = lo[e]; bvv[e + 8] = hi[e]; }
      bf[ni] = bvv;
    }
    #pragma unroll
    for (int mi = 0; mi < 2; ++mi)
      #pragma unroll
      for (int ni = 0; ni < 4; ++ni)
        acc[mi][ni] = __builtin_amdgcn_wmma_f32_16x16x32_f16(
            false, af[mi], false, bf[ni], (short)0, acc[mi][ni], false, false);
    sync_stage();
  }

  // Epilogue: C/D layout — VGPR r: rows r (lanes 0-15) / r+8 (lanes 16-31), col = lane%16
  #pragma unroll
  for (int ni = 0; ni < 4; ++ni) {
    const int ncol = n0 + wn * 64 + ni * 16 + (lane & 15);
    const float bv = bias[ncol];                 // hoisted: shared by all 16 rows
    #pragma unroll
    for (int mi = 0; mi < 2; ++mi)
      #pragma unroll
      for (int r = 0; r < 8; ++r) {
        const int mrow = m0 + wm * 32 + mi * 16 + ((lane >> 4) << 3) + r;
        const size_t gi = (size_t)mrow * N + ncol;
        float v = acc[mi][ni][r] + bv;
        if constexpr (HR) v += (float)resid[gi];
        if constexpr (ACT == 1) v = fmaxf(v, 0.f);
        if constexpr (ACT == 2) v = tanhf(v);
        C[gi] = (half_t)v;
      }
  }
}

// ---------------------------------------------------------------------------
// Attention: tiny 12x12 per (b, head). One thread per (b, head, query-row).
// ---------------------------------------------------------------------------
__global__ __launch_bounds__(256) void attn_kernel(
    const half_t* __restrict__ qkv, const int* __restrict__ glen,
    half_t* __restrict__ ao) {
  const int t = blockIdx.x * 256 + threadIdx.x;
  if (t >= kB * kNH * kS) return;
  const int b  = t / (kNH * kS);
  const int r  = t % (kNH * kS);
  const int h  = r / kS;
  const int qi = r % kS;
  const size_t base = (size_t)b * kS * kQKVN;
  const int ho = h * kHD;

  float sc[kS];
  #pragma unroll
  for (int j = 0; j < kS; ++j) sc[j] = 0.f;
  for (int d = 0; d < kHD; ++d) {
    const float qd = (float)qkv[base + (size_t)qi * kQKVN + ho + d];
    #pragma unroll
    for (int j = 0; j < kS; ++j)
      sc[j] += qd * (float)qkv[base + (size_t)j * kQKVN + kH + ho + d];
  }
  const float scale = 0.1020620726f;   // 1/sqrt(96)
  const int gl = glen[b];
  float mx = -1e30f;
  #pragma unroll
  for (int j = 0; j < kS; ++j) {
    sc[j] = (j < gl) ? sc[j] * scale : -1e9f;
    mx = fmaxf(mx, sc[j]);
  }
  float se = 0.f;
  #pragma unroll
  for (int j = 0; j < kS; ++j) { sc[j] = __expf(sc[j] - mx); se += sc[j]; }
  const float inv = 1.f / se;

  half_t* op = ao + (size_t)(b * kS + qi) * kH + ho;
  for (int d = 0; d < kHD; ++d) {
    float o = 0.f;
    #pragma unroll
    for (int j = 0; j < kS; ++j)
      o += sc[j] * (float)qkv[base + (size_t)j * kQKVN + 2 * kH + ho + d];
    op[d] = (half_t)(o * inv);
  }
}

// ---------------------------------------------------------------------------
// LayerNorm over H=768, one wave per row.
// ---------------------------------------------------------------------------
__global__ __launch_bounds__(256) void layernorm_kernel(
    const half_t* __restrict__ in, const float* __restrict__ g,
    const float* __restrict__ bta, half_t* __restrict__ out, int rows) {
  const int row = blockIdx.x * 8 + (threadIdx.x >> 5);
  if (row >= rows) return;
  const int lane = threadIdx.x & 31;
  const half_t* rp = in + (size_t)row * kH;

  float xv[kH / 32];
  float s = 0.f;
  #pragma unroll
  for (int tt = 0; tt < kH / 32; ++tt) { xv[tt] = (float)rp[lane + 32 * tt]; s += xv[tt]; }
  const float mu = wave_red_sum(s) * (1.f / kH);
  float v = 0.f;
  #pragma unroll
  for (int tt = 0; tt < kH / 32; ++tt) { const float d = xv[tt] - mu; v += d * d; }
  const float rs = rsqrtf(wave_red_sum(v) * (1.f / kH) + 1e-5f);
  half_t* op = out + (size_t)row * kH;
  #pragma unroll
  for (int tt = 0; tt < kH / 32; ++tt) {
    const int c = lane + 32 * tt;
    op[c] = (half_t)((xv[tt] - mu) * rs * g[c] + bta[c]);
  }
}

// ---------------------------------------------------------------------------
// Masked mean pool over valid sequence slots. One wave per batch element.
// ---------------------------------------------------------------------------
__global__ __launch_bounds__(256) void pool_kernel(
    const half_t* __restrict__ x, const int* __restrict__ glen,
    half_t* __restrict__ xg) {
  const int b = blockIdx.x * 8 + (threadIdx.x >> 5);
  if (b >= kB) return;
  const int lane = threadIdx.x & 31;
  const int gl = glen[b];
  const float inv = 1.f / (float)gl;
  #pragma unroll 4
  for (int tt = 0; tt < kH / 32; ++tt) {
    const int c = lane + 32 * tt;
    float s = 0.f;
    for (int j = 0; j < gl; ++j) s += (float)x[(size_t)(b * kS + j) * kH + c];
    xg[(size_t)b * kH + c] = (half_t)(s * inv);
  }
}

// ---------------------------------------------------------------------------
// 3-way gate: logits = [xd|xl|xg] @ Wg + bg, softmax, blend. Wave per element.
// ---------------------------------------------------------------------------
__global__ __launch_bounds__(256) void gate_kernel(
    const half_t* __restrict__ xd, const half_t* __restrict__ xl,
    const half_t* __restrict__ xg, const float* __restrict__ Wg,
    const float* __restrict__ bg, float* __restrict__ out) {
  const int b = blockIdx.x * 8 + (threadIdx.x >> 5);
  if (b >= kB) return;
  const int lane = threadIdx.x & 31;
  float l0 = 0.f, l1 = 0.f, l2 = 0.f;
  for (int i = lane; i < 3 * kH; i += 32) {
    float pf;
    if (i < kH)          pf = (float)xd[(size_t)b * kH + i];
    else if (i < 2 * kH) pf = (float)xl[(size_t)b * kH + (i - kH)];
    else                 pf = (float)xg[(size_t)b * kH + (i - 2 * kH)];
    l0 += pf * Wg[i * 3 + 0];
    l1 += pf * Wg[i * 3 + 1];
    l2 += pf * Wg[i * 3 + 2];
  }
  l0 = wave_red_sum(l0) + bg[0];
  l1 = wave_red_sum(l1) + bg[1];
  l2 = wave_red_sum(l2) + bg[2];
  const float mx = fmaxf(l0, fmaxf(l1, l2));
  const float e0 = __expf(l0 - mx), e1 = __expf(l1 - mx), e2 = __expf(l2 - mx);
  const float inv = 1.f / (e0 + e1 + e2);
  const float g0 = e0 * inv, g1 = e1 * inv, g2 = e2 * inv;
  #pragma unroll 4
  for (int tt = 0; tt < kH / 32; ++tt) {
    const int c = lane + 32 * tt;
    out[(size_t)b * kH + c] = g0 * (float)xd[(size_t)b * kH + c] +
                              g1 * (float)xl[(size_t)b * kH + c] +
                              g2 * (float)xg[(size_t)b * kH + c];
  }
}

// ---------------------------------------------------------------------------
// Host-side orchestration
// ---------------------------------------------------------------------------
extern "C" void kernel_launch(void* const* d_in, const int* in_sizes, int n_in,
                              void* d_out, int out_size, void* d_ws, size_t ws_size,
                              hipStream_t stream) {
  (void)in_sizes; (void)n_in; (void)out_size; (void)ws_size;
  const float* z    = (const float*)d_in[0];
  const float* buf  = (const float*)d_in[1];
  const int* bmask  = (const int*)d_in[2];
  const int* bidx   = (const int*)d_in[3];
  const float* Wq   = (const float*)d_in[4];
  const float* bq   = (const float*)d_in[5];
  const float* Wk   = (const float*)d_in[6];
  const float* bk   = (const float*)d_in[7];
  const float* Wv   = (const float*)d_in[8];
  const float* bv   = (const float*)d_in[9];
  const float* Wo   = (const float*)d_in[10];
  const float* bo   = (const float*)d_in[11];
  const float* ln1g = (const float*)d_in[12];
  const float* ln1b = (const float*)d_in[13];
  const float* W1   = (const float*)d_in[14];
  const float* b1   = (const float*)d_in[15];
  const float* W2   = (const float*)d_in[16];
  const float* b2   = (const float*)d_in[17];
  const float* ln2g = (const float*)d_in[18];
  const float* ln2b = (const float*)d_in[19];
  const float* Wd   = (const float*)d_in[20];
  const float* bd   = (const float*)d_in[21];
  const float* Wl   = (const float*)d_in[22];
  const float* bl   = (const float*)d_in[23];
  const float* Wg   = (const float*)d_in[24];
  const float* bg   = (const float*)d_in[25];
  float* out = (float*)d_out;

  char* ws = (char*)d_ws;
  size_t off = 0;
  auto alloc = [&](size_t bytes) -> char* {
    char* p = ws + off;
    off += (bytes + 255) & ~((size_t)255);
    return p;
  };

  // All weight copies are transposed: [N][K], f16.
  half_t* wqkv16  = (half_t*)alloc((size_t)kQKVN * kH * 2);
  float*  bqkv    = (float*)alloc((size_t)kQKVN * 4);
  half_t* wo16    = (half_t*)alloc((size_t)kH * kH * 2);
  half_t* w116    = (half_t*)alloc((size_t)kFF * kH * 2);
  half_t* w216    = (half_t*)alloc((size_t)kH * kFF * 2);
  half_t* wd16    = (half_t*)alloc((size_t)kH * kH * 2);
  half_t* wl16    = (half_t*)alloc((size_t)kH * kH * 2);
  half_t* z16     = (half_t*)alloc((size_t)kB * kH * 2);
  half_t* lpool16 = (half_t*)alloc((size_t)kB * kH * 2);
  int*    glen    = (int*)alloc((size_t)kB * 4);
  half_t* gseq16  = (half_t*)alloc((size_t)kNROW * kH * 2);     // reused as x2n
  half_t* qkv16   = (half_t*)alloc((size_t)kNROW * kQKVN * 2);  // reused as x1raw, then h1
  half_t* ao16    = (half_t*)alloc((size_t)kNROW * kH * 2);     // reused as x2raw
  half_t* x1n     = (half_t*)alloc((size_t)kNROW * kH * 2);
  half_t* xg16    = (half_t*)alloc((size_t)kB * kH * 2);
  half_t* xd16    = (half_t*)alloc((size_t)kB * kH * 2);
  half_t* xl16    = (half_t*)alloc((size_t)kB * kH * 2);

  half_t* x1raw = qkv16;   // after attention, the QKV buffer is free
  half_t* h1    = qkv16;   // after LN1, x1raw is free
  half_t* x2raw = ao16;    // after the Wo GEMM, ao is free
  half_t* x2n   = gseq16;  // after the Wo GEMM (residual read), gseq is free

  const dim3 blk(256);

  // 1) weight conversion (transpose to [N][K] f16)
  pack_qkv_t_kernel<<<(kQKVN * kH + 255) / 256, blk, 0, stream>>>(Wq, Wk, Wv, wqkv16);
  pack_bias3_kernel<<<(kQKVN + 255) / 256, blk, 0, stream>>>(bq, bk, bv, bqkv);
  cvt_transpose_kernel<<<(kH * kH + 255) / 256, blk, 0, stream>>>(Wo, wo16, kH, kH);
  cvt_transpose_kernel<<<(kH * kFF + 255) / 256, blk, 0, stream>>>(W1, w116, kH, kFF);
  cvt_transpose_kernel<<<(kFF * kH + 255) / 256, blk, 0, stream>>>(W2, w216, kFF, kH);
  cvt_transpose_kernel<<<(kH * kH + 255) / 256, blk, 0, stream>>>(Wd, wd16, kH, kH);
  cvt_transpose_kernel<<<(kH * kH + 255) / 256, blk, 0, stream>>>(Wl, wl16, kH, kH);

  // 2) sequence gather + local pool
  build_seq_kernel<<<kB / 8, blk, 0, stream>>>(z, buf, bmask, bidx, gseq16, z16, lpool16, glen);

  // 3) QKV: (98304 x 768) @ (768 x 2304)
  gemm_f16_kernel<false, 0><<<dim3(kQKVN / 128, kNROW / 128), blk, 0, stream>>>(
      gseq16, wqkv16, bqkv, nullptr, qkv16, kNROW, kQKVN, kH);

  // 4) attention (12x12 per head)
  attn_kernel<<<(kB * kNH * kS) / 256, blk, 0, stream>>>(qkv16, glen, ao16);

  // 5) x1raw = ao @ Wo + bo + gseq
  gemm_f16_kernel<true, 0><<<dim3(kH / 128, kNROW / 128), blk, 0, stream>>>(
      ao16, wo16, bo, gseq16, x1raw, kNROW, kH, kH);

  // 6) LN1
  layernorm_kernel<<<kNROW / 8, blk, 0, stream>>>(x1raw, ln1g, ln1b, x1n, kNROW);

  // 7) h1 = relu(x1n @ W1 + b1)
  gemm_f16_kernel<false, 1><<<dim3(kFF / 128, kNROW / 128), blk, 0, stream>>>(
      x1n, w116, b1, nullptr, h1, kNROW, kFF, kH);

  // 8) x2raw = h1 @ W2 + b2 + x1n
  gemm_f16_kernel<true, 0><<<dim3(kH / 128, kNROW / 128), blk, 0, stream>>>(
      h1, w216, b2, x1n, x2raw, kNROW, kH, kFF);

  // 9) LN2
  layernorm_kernel<<<kNROW / 8, blk, 0, stream>>>(x2raw, ln2g, ln2b, x2n, kNROW);

  // 10) global pool
  pool_kernel<<<kB / 8, blk, 0, stream>>>(x2n, glen, xg16);

  // 11) x_d = tanh(z @ Wd + bd) ; x_l = tanh(lpool @ Wl + bl)
  gemm_f16_kernel<false, 2><<<dim3(kH / 128, kB / 128), blk, 0, stream>>>(
      z16, wd16, bd, nullptr, xd16, kB, kH, kH);
  gemm_f16_kernel<false, 2><<<dim3(kH / 128, kB / 128), blk, 0, stream>>>(
      lpool16, wl16, bl, nullptr, xl16, kB, kH, kH);

  // 12) gate + blend
  gate_kernel<<<kB / 8, blk, 0, stream>>>(xd16, xl16, xg16, Wg, bg, out);
}
